// OpticalFC_28999619182926
// MI455X (gfx1250) — 8-bit matrix operand order repaired and checked by reading, compiled, not run on hardware
//
#include <hip/hip_runtime.h>
#include <hip/hip_bf16.h>


typedef int   v8i __attribute__((ext_vector_type(8)));
typedef int   v4i __attribute__((ext_vector_type(4)));
typedef int   v2i __attribute__((ext_vector_type(2)));
typedef float v4f __attribute__((ext_vector_type(4)));

typedef __attribute__((address_space(1))) v4i* gv4i_p;
typedef __attribute__((address_space(3))) v4i* lv4i_p;

#define B_DIM   4096
#define IN_DIM  4096
#define OUT_DIM 4096
#define INV_S   (1.0f / 4032.25f)
#define KSTEPS  (IN_DIM / 64)

__device__ __forceinline__ void async_copy16(const signed char* g, signed char* l)
{
#if __has_builtin(__builtin_amdgcn_global_load_async_to_lds_b128)
    __builtin_amdgcn_global_load_async_to_lds_b128((gv4i_p)g, (lv4i_p)l, 0, 0);
#else
    *(v4i*)l = *(const v4i*)g;
#endif
}

__device__ __forceinline__ void wait_async_done()
{
#if __has_builtin(__builtin_amdgcn_global_load_async_to_lds_b128)
#  if __has_builtin(__builtin_amdgcn_s_wait_asynccnt)
    __builtin_amdgcn_s_wait_asynccnt(0);
#  else
    asm volatile("s_wait_asynccnt 0" ::: "memory");
#  endif
#endif
}


__global__ __launch_bounds__(256) void quant_pack_kernel(
    const float* __restrict__ src, unsigned int* __restrict__ dst,
    float scale, float clip, int n4)
{
    int i = blockIdx.x * blockDim.x + threadIdx.x;
    if (i >= n4) return;
    v4f v = __builtin_nontemporal_load((const v4f*)src + i);
    int a = (int)rintf(fminf(fmaxf(v.x, -clip), clip) * scale);
    int b = (int)rintf(fminf(fmaxf(v.y, -clip), clip) * scale);
    int c = (int)rintf(fminf(fmaxf(v.z, -clip), clip) * scale);
    int d = (int)rintf(fminf(fmaxf(v.w, -clip), clip) * scale);
    dst[i] = (unsigned int)(a & 255) | ((unsigned int)(b & 255) << 8) |
             ((unsigned int)(c & 255) << 16) | ((unsigned int)(d & 255) << 24);
}

__global__ __launch_bounds__(256) void quant_bias_kernel(
    const float* __restrict__ bias, float* __restrict__ biasF, int n)
{
    int i = blockIdx.x * blockDim.x + threadIdx.x;
    if (i >= n) return;
    float bi = rintf(fminf(fmaxf(bias[i], -1.0f), 1.0f) * 127.0f);
    biasF[i] = 32.0f * bi * INV_S;
}


__global__ __launch_bounds__(256) void optical_gemm_kernel(
    const signed char* __restrict__ xq,
    const signed char* __restrict__ wq,
    const float* __restrict__ biasF,
    const float* __restrict__ noise,
    float* __restrict__ out)
{
    __shared__ __align__(16) signed char sA[2][64 * 64];
    __shared__ __align__(16) signed char sB[2][256 * 64];

    const int tid   = threadIdx.x;
    const int lane  = tid & 31;
    const int l16   = lane & 15;
    const int hi    = lane >> 4;
    const int wave  = tid >> 5;
    const int waveM = wave >> 2;
    const int waveN = wave & 3;

    const int tileN = blockIdx.x * 256;
    const int tileM = blockIdx.y * 64;

    const int srow = tid >> 2;
    const int scol = (tid & 3) * 16;

    const size_t aGBase = (size_t)(tileM + srow) * IN_DIM + scol;
    const size_t bGBase = (size_t)(tileN + srow) * IN_DIM + scol;

    v8i acc[2][4] = {};

    async_copy16(&xq[aGBase], &sA[0][srow * 64 + scol]);
#pragma unroll
    for (int r = 0; r < 4; ++r)
        async_copy16(&wq[bGBase + (size_t)(64 * r) * IN_DIM],
                     &sB[0][(srow + 64 * r) * 64 + scol]);
    wait_async_done();
    __syncthreads();

    for (int kt = 0; kt < KSTEPS; ++kt) {
        const int cur = kt & 1;
        const int nxt = cur ^ 1;
        const size_t kOffN = (size_t)(kt + 1) * 64;

        if (kt + 1 < KSTEPS) {
            async_copy16(&xq[aGBase + kOffN], &sA[nxt][srow * 64 + scol]);
#pragma unroll
            for (int r = 0; r < 4; ++r)
                async_copy16(&wq[bGBase + (size_t)(64 * r) * IN_DIM + kOffN],
                             &sB[nxt][(srow + 64 * r) * 64 + scol]);
            if (kt + 2 < KSTEPS) {
                __builtin_prefetch(&xq[aGBase + kOffN + 64], 0, 1);
                __builtin_prefetch(&wq[bGBase + kOffN + 64], 0, 1);
            }
        }

        v8i afrag[2], bfrag[4];
#pragma unroll
        for (int i = 0; i < 2; ++i) {
            const int m = waveM * 32 + i * 16 + l16;
            v2i* ap = (v2i*)&afrag[i];
#pragma unroll
            for (int p = 0; p < 4; ++p)
                ap[p] = *(const v2i*)&sA[cur][m * 64 + p * 16 + hi * 8];
        }
#pragma unroll
        for (int j = 0; j < 4; ++j) {
            const int o = waveN * 64 + j * 16 + l16;
            v2i* bp = (v2i*)&bfrag[j];
#pragma unroll
            for (int p = 0; p < 4; ++p)
                bp[p] = *(const v2i*)&sB[cur][o * 64 + p * 16 + hi * 8];
        }

#pragma unroll
        for (int i = 0; i < 2; ++i)
#pragma unroll
            for (int j = 0; j < 4; ++j)
                acc[i][j] = __builtin_amdgcn_wmma_i32_16x16x64_iu8(
                     true, afrag[i],
                     true, bfrag[j],
                    acc[i][j],  false,  false);

        wait_async_done();
        __syncthreads();
    }

#pragma unroll
    for (int i = 0; i < 2; ++i) {
#pragma unroll
        for (int j = 0; j < 4; ++j) {
            const int n = tileN + waveN * 64 + j * 16 + l16;
            const float bn = biasF[n];
#pragma unroll
            for (int r = 0; r < 8; ++r) {
                const int m = tileM + waveM * 32 + i * 16 + r + hi * 8;
                const size_t idx = (size_t)m * OUT_DIM + n;
                float nz = __builtin_nontemporal_load(&noise[idx]);
                float v  = fmaf((float)acc[i][j][r], INV_S,
                                fmaf(0.01f, nz, bn));
                __builtin_nontemporal_store(v, &out[idx]);
            }
        }
    }
}


extern "C" void kernel_launch(void* const* d_in, const int* in_sizes, int n_in,
                              void* d_out, int out_size, void* d_ws, size_t ws_size,
                              hipStream_t stream)
{
    const float* x     = (const float*)d_in[0];
    const float* w     = (const float*)d_in[1];
    const float* bias  = (const float*)d_in[2];
    const float* noise = (const float*)d_in[3];
    float* out = (float*)d_out;

    signed char* xq   = (signed char*)d_ws;
    signed char* wq   = (signed char*)d_ws + (size_t)16 * 1024 * 1024;
    float* biasF      = (float*)((char*)d_ws + (size_t)32 * 1024 * 1024);

    const int n4 = (B_DIM * IN_DIM) / 4;
    quant_pack_kernel<<<(n4 + 255) / 256, 256, 0, stream>>>(
        x, (unsigned int*)xq, 31.75f, 4.0f, n4);
    quant_pack_kernel<<<(n4 + 255) / 256, 256, 0, stream>>>(
        w, (unsigned int*)wq, 127.0f, 1.0f, n4);
    quant_bias_kernel<<<(OUT_DIM + 255) / 256, 256, 0, stream>>>(
        bias, biasF, OUT_DIM);

    dim3 grid(OUT_DIM / 256, B_DIM / 64);
    optical_gemm_kernel<<<grid, 256, 0, stream>>>(xq, wq, biasF, noise, out);
}
